// EMTransformerEncoder_56959856279619
// MI455X (gfx1250) — matrix operand order repaired and checked by reading, not compiled, not run on hardware
//
#include <hip/hip_runtime.h>
#include <hip/hip_bf16.h>
#include <math.h>

typedef __attribute__((ext_vector_type(16))) _Float16 v16h;
typedef __attribute__((ext_vector_type(8)))  _Float16 v8h;
typedef __attribute__((ext_vector_type(8)))  float    v8f;
typedef __attribute__((ext_vector_type(4)))  float    v4f;
typedef __attribute__((ext_vector_type(4)))  int      v4i;

#define D_MODEL  256
#define D_FF     1024
#define N_LAYERS 6
#define B_BATCH  4
#define HF_      256
#define WF_      256
#define LLEV     4
#define LDS_PAD  40   // padded K-stride (halves): 80B row stride -> conflict-free, 16B aligned

// -------- gfx1250 async global->LDS copy (guarded; falls back to reg-staged sync copy) --------
#if defined(__HIP_DEVICE_COMPILE__) && __has_builtin(__builtin_amdgcn_global_load_async_to_lds_b128)
#define USE_ASYNC_LDS 1
#else
#define USE_ASYNC_LDS 0
#endif

__device__ __forceinline__ void async_cp_b128(const _Float16* g, _Float16* l) {
#if USE_ASYNC_LDS
    // signature (probe-confirmed arity 4; param1 is int4 addrspace(1)*):
    //   (global int4* src, LDS int4* dst, imm offset, imm cpol)
    __builtin_amdgcn_global_load_async_to_lds_b128(
        (__attribute__((address_space(1))) v4i*)g,
        (__attribute__((address_space(3))) v4i*)l, 0, 0);
#else
    (void)g; (void)l;
#endif
}
__device__ __forceinline__ void async_wait_all() {
#if USE_ASYNC_LDS
    asm volatile("s_wait_asynccnt 0x0" ::: "memory");
#endif
}

// ---------------- device helpers ----------------
__device__ __forceinline__ float gelu_tanh(float x) {
    const float k0 = 0.7978845608028654f; // sqrt(2/pi)
    float t = tanhf(k0 * (x + 0.044715f * x * x * x));
    return 0.5f * x * (1.0f + t);
}
__device__ __forceinline__ float sigmoidf(float x) { return 1.0f / (1.0f + expf(-x)); }

__device__ __forceinline__ v16h pack16(v8h lo, v8h hi) {
    v16h o;
#pragma unroll
    for (int j = 0; j < 8; ++j) { o[j] = lo[j]; o[j + 8] = hi[j]; }
    return o;
}

// ---------------- small utility kernels ----------------
__global__ void lin_kernel(const int* __restrict__ idx4, int* __restrict__ lin, int n) {
    int i = blockIdx.x * blockDim.x + threadIdx.x;
    if (i >= n) return;
    const int* r = idx4 + (size_t)i * 4;
    lin[i] = ((r[0] * HF_ + r[1]) * WF_ + r[2]) * LLEV + r[3];
}

__global__ void clear_flags_kernel(unsigned char* __restrict__ f, int n) {
    int i = blockIdx.x * blockDim.x + threadIdx.x;
    if (i < n) f[i] = 0;
}

// converts this layer's weights to f16 AND transposes them to [n][k] so that
// both GEMM operands stage as contiguous row-major 128x32 tiles (b128 async copies)
__global__ void wconv_kernel(const float* __restrict__ W1, const float* __restrict__ W2,
                             int layer, _Float16* __restrict__ W1t, _Float16* __restrict__ W2t) {
    int i = blockIdx.x * blockDim.x + threadIdx.x;
    if (i < D_MODEL * D_FF) {
        int k = i / D_FF, n = i % D_FF;          // W1[layer][k][n]
        W1t[(size_t)n * D_MODEL + k] = (_Float16)W1[(size_t)layer * D_MODEL * D_FF + i];
    } else if (i < 2 * D_MODEL * D_FF) {
        int j = i - D_MODEL * D_FF;
        int k = j / D_MODEL, n = j % D_MODEL;    // W2[layer][k][n]
        W2t[(size_t)n * D_FF + k] = (_Float16)W2[(size_t)layer * D_FF * D_MODEL + j];
    }
}

// ---------------- gather + score (one wave32 per query row) ----------------
__global__ __launch_bounds__(256) void gather_score_kernel(
    const float* __restrict__ vals, const float* __restrict__ pos_values,
    const float* __restrict__ salience, const int* __restrict__ ij_level,
    const int* __restrict__ subset_idx, const float* __restrict__ w_score,
    const float* __restrict__ b_score, const int* __restrict__ lin,
    int n_sparse, int T, int Kq, int layer,
    int* __restrict__ pos_out, float* __restrict__ coef_out,
    _Float16* __restrict__ X, unsigned char* __restrict__ flags, int set_flags)
{
    int wave = (blockIdx.x * blockDim.x + threadIdx.x) >> 5;
    int lane = threadIdx.x & 31;
    int BK = B_BATCH * Kq;
    if (wave >= BK) return;
    int b  = wave / Kq;
    int kk = wave - b * Kq;
    int sel = subset_idx[((size_t)b * N_LAYERS + layer) * Kq + kk];
    const int* ij = ij_level + ((size_t)b * T + sel) * 4;
    int qlin = ((ij[0] * HF_ + ij[1]) * WF_ + ij[2]) * LLEV + ij[3];

    // lower_bound over sorted lin[] (uniform across lanes; exact match guaranteed)
    int lo = 0, hi = n_sparse;
    while (lo < hi) { int mid = (lo + hi) >> 1; if (lin[mid] < qlin) lo = mid + 1; else hi = mid; }
    int pos = lo;

    // 8 contiguous columns per lane
    const float* qp  = vals       + (size_t)pos * D_MODEL + lane * 8;
    const float* pep = pos_values + (size_t)pos * D_MODEL + lane * 8;
    v4f q0 = *(const v4f*)(qp);      v4f q1 = *(const v4f*)(qp + 4);
    v4f p0 = *(const v4f*)(pep);     v4f p1 = *(const v4f*)(pep + 4);
    const float* wp = w_score + lane * 8;
    float partial = 0.0f;
    v8h xv;
#pragma unroll
    for (int j = 0; j < 4; ++j) {
        partial += q0[j] * wp[j];
        partial += q1[j] * wp[4 + j];
        xv[j]     = (_Float16)(q0[j] + p0[j]);
        xv[4 + j] = (_Float16)(q1[j] + p1[j]);
    }
    *(v8h*)(X + (size_t)wave * D_MODEL + lane * 8) = xv;

#pragma unroll
    for (int off = 16; off > 0; off >>= 1) partial += __shfl_xor(partial, off, 32);
    if (lane == 0) {
        float prob = sigmoidf(partial + b_score[0]);
        pos_out[wave]  = pos;
        coef_out[wave] = prob * salience[(size_t)b * T + sel];
        if (set_flags) flags[pos] = 1;
    }
}

// ---------------- double-buffered WMMA 128x128 tile core ----------------
// C(128x128) += A(rows m0..m0+127 of [M][lda]) * B^T(rows n0..n0+127 of [N][Ktot])
__device__ __forceinline__ void wmma_mm_tile(
    const _Float16* __restrict__ Ag, int lda,
    const _Float16* __restrict__ Bt,             // transposed weights [N][Ktot]
    int m0, int n0, int Ktot,
    _Float16* lA0, _Float16* lB0, _Float16* lA1, _Float16* lB1,
    v8f acc[8])
{
    const int tid  = threadIdx.x;
    const int wave = tid >> 5;
    const int lane = tid & 31;
    const int r    = lane & 15;
    const int hi   = lane >> 4;
    // staging: 512 chunks of 8 halves (16B) per 128x32 tile; 2 chunks/thread
    const int crr = tid >> 2;          // 0..63 (and +64 for second chunk)
    const int ckc = (tid & 3) * 8;     // 0,8,16,24

#pragma unroll
    for (int i = 0; i < 8; ++i)
        acc[i] = (v8f){0.f, 0.f, 0.f, 0.f, 0.f, 0.f, 0.f, 0.f};

#if !USE_ASYNC_LDS
    v8h ra0{}, ra1{}, rb0{}, rb1{};
#endif

#define STAGE_ASYNC(kk, dA, dB)                                                          \
    do {                                                                                 \
        async_cp_b128(Ag + (size_t)(m0 + crr)      * lda  + (kk) + ckc, dA + crr * LDS_PAD + ckc);        \
        async_cp_b128(Ag + (size_t)(m0 + crr + 64) * lda  + (kk) + ckc, dA + (crr + 64) * LDS_PAD + ckc); \
        async_cp_b128(Bt + (size_t)(n0 + crr)      * Ktot + (kk) + ckc, dB + crr * LDS_PAD + ckc);        \
        async_cp_b128(Bt + (size_t)(n0 + crr + 64) * Ktot + (kk) + ckc, dB + (crr + 64) * LDS_PAD + ckc); \
    } while (0)
#define LOAD_REGS(kk)                                                                    \
    do {                                                                                 \
        ra0 = *(const v8h*)(Ag + (size_t)(m0 + crr)      * lda  + (kk) + ckc);           \
        ra1 = *(const v8h*)(Ag + (size_t)(m0 + crr + 64) * lda  + (kk) + ckc);           \
        rb0 = *(const v8h*)(Bt + (size_t)(n0 + crr)      * Ktot + (kk) + ckc);           \
        rb1 = *(const v8h*)(Bt + (size_t)(n0 + crr + 64) * Ktot + (kk) + ckc);           \
    } while (0)
#define STORE_REGS(dA, dB)                                                               \
    do {                                                                                 \
        *(v8h*)(dA + crr * LDS_PAD + ckc)        = ra0;                                  \
        *(v8h*)(dA + (crr + 64) * LDS_PAD + ckc) = ra1;                                  \
        *(v8h*)(dB + crr * LDS_PAD + ckc)        = rb0;                                  \
        *(v8h*)(dB + (crr + 64) * LDS_PAD + ckc) = rb1;                                  \
    } while (0)

    // prologue: stage first K-tile
#if USE_ASYNC_LDS
    STAGE_ASYNC(0, lA0, lB0);
    async_wait_all();
#else
    LOAD_REGS(0);
    STORE_REGS(lA0, lB0);
#endif
    __syncthreads();

    _Float16 *cA = lA0, *cB = lB0, *nA = lA1, *nB = lB1;
    for (int k0 = 0; k0 < Ktot; k0 += 32) {
        const bool notlast = (k0 + 32 < Ktot);
        // issue next tile's copies before computing on the current one
#if USE_ASYNC_LDS
        if (notlast) STAGE_ASYNC(k0 + 32, nA, nB);
#else
        if (notlast) LOAD_REGS(k0 + 32);
#endif
        // A fragment 16x32: lane(0-15)=row, K {hi*8..+7} then {16+hi*8..+7}
        const _Float16* ap = cA + (wave * 16 + r) * LDS_PAD;
        v16h a = pack16(*(const v8h*)(ap + hi * 8), *(const v8h*)(ap + 16 + hi * 8));
#pragma unroll
        for (int ns = 0; ns < 8; ++ns) {
            // B fragment 32x16: lane col c=r, K {hi*8..+7} then {16+hi*8..+7} (matches A)
            const _Float16* bp = cB + (ns * 16 + r) * LDS_PAD;
            v16h bb = pack16(*(const v8h*)(bp + hi * 8), *(const v8h*)(bp + 16 + hi * 8));
            acc[ns] = __builtin_amdgcn_wmma_f32_16x16x32_f16(
                false, a, false, bb, (short)0, acc[ns], false, false);
        }
        if (notlast) {
#if USE_ASYNC_LDS
            async_wait_all();
#else
            STORE_REGS(nA, nB);
#endif
            __syncthreads();
            _Float16* t;
            t = cA; cA = nA; nA = t;
            t = cB; cB = nB; nB = t;
        }
    }
#undef STAGE_ASYNC
#undef LOAD_REGS
#undef STORE_REGS
}

// ---------------- FFN layer 1: H = gelu(X @ W1 + b1) ----------------
__global__ __launch_bounds__(256) void ffn1_kernel(
    const _Float16* __restrict__ X, const _Float16* __restrict__ W1t,
    const float* __restrict__ b1, int layer, _Float16* __restrict__ H)
{
    __shared__ __align__(32) _Float16 smem[4 * 128 * LDS_PAD];
    _Float16* lA0 = smem;
    _Float16* lB0 = smem + 128 * LDS_PAD;
    _Float16* lA1 = smem + 2 * 128 * LDS_PAD;
    _Float16* lB1 = smem + 3 * 128 * LDS_PAD;
    const int m0 = blockIdx.x * 128;
    const int n0 = blockIdx.y * 128;
    v8f acc[8];
    wmma_mm_tile(X, D_MODEL, W1t, m0, n0, D_MODEL, lA0, lB0, lA1, lB1, acc);

    const int wave = threadIdx.x >> 5;
    const int lane = threadIdx.x & 31;
    const int c = lane & 15, hi = lane >> 4;
    const float* bias = b1 + (size_t)layer * D_FF;
#pragma unroll
    for (int ns = 0; ns < 8; ++ns) {
        int col = n0 + ns * 16 + c;
        float bv = bias[col];
#pragma unroll
        for (int v = 0; v < 8; ++v) {
            int row = m0 + wave * 16 + hi * 8 + v;
            H[(size_t)row * D_FF + col] = (_Float16)gelu_tanh(acc[ns][v] + bv);
        }
    }
}

// ---------------- FFN layer 2 + scaled scatter-add: vals[pos[r]] += (H@W2+b2)*coef[r] ----------------
__global__ __launch_bounds__(256) void ffn2_scatter_kernel(
    const _Float16* __restrict__ H, const _Float16* __restrict__ W2t,
    const float* __restrict__ b2, int layer,
    const int* __restrict__ pos, const float* __restrict__ coef,
    float* __restrict__ vals)
{
    __shared__ __align__(32) _Float16 smem[4 * 128 * LDS_PAD];
    _Float16* lA0 = smem;
    _Float16* lB0 = smem + 128 * LDS_PAD;
    _Float16* lA1 = smem + 2 * 128 * LDS_PAD;
    _Float16* lB1 = smem + 3 * 128 * LDS_PAD;
    const int m0 = blockIdx.x * 128;
    const int n0 = blockIdx.y * 128;
    v8f acc[8];
    wmma_mm_tile(H, D_FF, W2t, m0, n0, D_FF, lA0, lB0, lA1, lB1, acc);

    const int wave = threadIdx.x >> 5;
    const int lane = threadIdx.x & 31;
    const int c = lane & 15, hi = lane >> 4;
    const float* bias = b2 + (size_t)layer * D_MODEL;
    int posr[8]; float coefr[8];
#pragma unroll
    for (int v = 0; v < 8; ++v) {
        int row = m0 + wave * 16 + hi * 8 + v;
        posr[v]  = pos[row];
        coefr[v] = coef[row];
    }
#pragma unroll
    for (int ns = 0; ns < 8; ++ns) {
        int col = n0 + ns * 16 + c;
        float bv = bias[col];
#pragma unroll
        for (int v = 0; v < 8; ++v) {
            float* p = vals + (size_t)posr[v] * D_MODEL + col;   // pos unique per layer: race-free
            *p += (acc[ns][v] + bv) * coefr[v];
        }
    }
}

// ---------------- background Fourier PE for non-member rows ----------------
__global__ __launch_bounds__(256) void background_kernel(
    const int* __restrict__ idx4, const int* __restrict__ shapes,
    const float* __restrict__ FW, const unsigned char* __restrict__ flags,
    float* __restrict__ out, int n)
{
    int row = blockIdx.x;
    if (row >= n || flags[row]) return;
    const int* r = idx4 + (size_t)row * 4;
    int lev = r[3];
    float Hs = (float)shapes[lev * 2 + 0];
    float Ws = (float)shapes[lev * 2 + 1];
    float x0 = ((float)r[1] + 0.5f) / Hs;
    float x1 = ((float)r[2] + 0.5f) / Ws;
    float x2 = (float)lev * (1.0f / (float)(LLEV - 1));
    int col = threadIdx.x;
    int f = col & 127;
    float proj = 6.283185307179586f *
        (x0 * FW[f] + x1 * FW[128 + f] + x2 * FW[256 + f]);
    out[(size_t)row * D_MODEL + col] = (col < 128) ? sinf(proj) : cosf(proj);
}

// ---------------- host launcher ----------------
extern "C" void kernel_launch(void* const* d_in, const int* in_sizes, int n_in,
                              void* d_out, int out_size, void* d_ws, size_t ws_size,
                              hipStream_t stream)
{
    (void)n_in; (void)out_size; (void)ws_size;
    const int*   sparse_indices = (const int*)  d_in[0];
    const float* sparse_values  = (const float*)d_in[1];
    const float* pos_values     = (const float*)d_in[2];
    const int*   spatial_shapes = (const int*)  d_in[3];
    const float* salience       = (const float*)d_in[4];
    const int*   ij_level       = (const int*)  d_in[5];
    // d_in[6] = xy_pos (unused by reference)
    const int*   subset_idx     = (const int*)  d_in[7];
    const float* w_score        = (const float*)d_in[8];
    const float* b_score        = (const float*)d_in[9];
    const float* W1             = (const float*)d_in[10];
    const float* b1             = (const float*)d_in[11];
    const float* W2             = (const float*)d_in[12];
    const float* b2             = (const float*)d_in[13];
    const float* fourier_W      = (const float*)d_in[14];

    const int Nrows = in_sizes[1] / D_MODEL;                 // 348160
    const int T     = in_sizes[4] / B_BATCH;                 // 20000
    const int Kq    = in_sizes[7] / (B_BATCH * N_LAYERS);    // 8192
    const int BK    = B_BATCH * Kq;                          // 32768

    // workspace carve-up (~87 MB total)
    char* wsp = (char*)d_ws;
    auto wsalloc = [&](size_t bytes) -> char* {
        char* p = wsp; wsp += (bytes + 255) & ~(size_t)255; return p;
    };
    int*           lin   = (int*)          wsalloc((size_t)Nrows * 4);
    unsigned char* flags = (unsigned char*)wsalloc((size_t)Nrows);
    int*           pos   = (int*)          wsalloc((size_t)BK * 4);
    float*         coef  = (float*)        wsalloc((size_t)BK * 4);
    _Float16*      Xh    = (_Float16*)     wsalloc((size_t)BK * D_MODEL * 2);
    _Float16*      Hh    = (_Float16*)     wsalloc((size_t)BK * D_FF * 2);
    _Float16*      W1t   = (_Float16*)     wsalloc((size_t)D_MODEL * D_FF * 2);
    _Float16*      W2t   = (_Float16*)     wsalloc((size_t)D_FF * D_MODEL * 2);

    float* vals = (float*)d_out;
    (void)hipMemcpyAsync(vals, sparse_values, (size_t)Nrows * D_MODEL * sizeof(float),
                         hipMemcpyDeviceToDevice, stream);

    lin_kernel<<<(Nrows + 255) / 256, 256, 0, stream>>>(sparse_indices, lin, Nrows);
    clear_flags_kernel<<<(Nrows + 255) / 256, 256, 0, stream>>>(flags, Nrows);

    for (int layer = 0; layer < N_LAYERS; ++layer) {
        wconv_kernel<<<(2 * D_MODEL * D_FF + 255) / 256, 256, 0, stream>>>(
            W1, W2, layer, W1t, W2t);
        gather_score_kernel<<<(BK * 32 + 255) / 256, 256, 0, stream>>>(
            vals, pos_values, salience, ij_level, subset_idx, w_score, b_score,
            lin, Nrows, T, Kq, layer, pos, coef, Xh, flags,
            (layer == N_LAYERS - 1) ? 1 : 0);
        dim3 g1(BK / 128, D_FF / 128);
        ffn1_kernel<<<g1, 256, 0, stream>>>(Xh, W1t, b1, layer, Hh);
        dim3 g2(BK / 128, D_MODEL / 128);
        ffn2_scatter_kernel<<<g2, 256, 0, stream>>>(Hh, W2t, b2, layer, pos, coef, vals);
    }

    background_kernel<<<Nrows, 256, 0, stream>>>(
        sparse_indices, spatial_shapes, fourier_W, flags, vals, Nrows);
}
